// GAT_2_64037962383538
// MI455X (gfx1250) — matrix operand order repaired and checked by reading, not compiled, not run on hardware
//
#include <hip/hip_runtime.h>
#include <hip/hip_bf16.h>

// ---------------------------------------------------------------------------
// GAT (3 GATConv + 2 Linear, BN+ReLU, mean-pool, classifier) for gfx1250.
// GEMMs use v_wmma_f32_16x16x32_f16 (wave32 WMMA). Edge softmax/aggregation
// via L2-resident float atomics (N*128*4B = 51MB << 192MB L2).
// ---------------------------------------------------------------------------

typedef __attribute__((ext_vector_type(16))) _Float16 v16h;
typedef __attribute__((ext_vector_type(8)))  float    v8f;

#define NEG_SLOPE 0.2f
#define EPS_BN 1e-5f

__device__ __forceinline__ float lrelu(float v) { return v > 0.f ? v : NEG_SLOPE * v; }

__device__ __forceinline__ void atomicMaxF(float* addr, float val) {
  // valid when buffer initialized to -FLT_MAX-ish
  if (val >= 0.f) atomicMax((int*)addr, __float_as_int(val));
  else            atomicMin((unsigned int*)addr, __float_as_uint(val));
}

__device__ __forceinline__ void edge_endpoints(const int* __restrict__ ei, int E,
                                               int e, int& s, int& d) {
  if (e < E) { s = ei[e]; d = ei[E + e]; }
  else       { s = e - E; d = e - E; }          // self loop
}

// ------------------------------ fills --------------------------------------
__global__ void fill_kernel(float* __restrict__ p, float v, int n) {
  int i = blockIdx.x * blockDim.x + threadIdx.x;
  if (i < n) p[i] = v;
}

__global__ void fill_bias_kernel(float* __restrict__ agg, const float* __restrict__ bias,
                                 int F, int ntot) {
  int i = blockIdx.x * blockDim.x + threadIdx.x;
  if (i < ntot) agg[i] = bias[i % F];
}

// -------------------------- f16 staging ------------------------------------
// A: [M, Kp] f16 row-major, zero-padded K
__global__ void conv_a_f16(const float* __restrict__ src, int srcStride, int validK,
                           _Float16* __restrict__ dst, int Kp, int M) {
  int i = blockIdx.x * blockDim.x + threadIdx.x;
  if (i >= M * Kp) return;
  int n = i / Kp, k = i % Kp;
  dst[i] = (k < validK) ? (_Float16)src[(size_t)n * srcStride + k] : (_Float16)0.f;
}

// W [K, Nc] row-major -> Wt [Ncp, Kp] (column n's K-vector contiguous), zero pad
__global__ void conv_w_f16(const float* __restrict__ W, int K, int Nc,
                           _Float16* __restrict__ Wt, int Kp, int Ncp) {
  int i = blockIdx.x * blockDim.x + threadIdx.x;
  if (i >= Ncp * Kp) return;
  int n = i / Kp, k = i % Kp;
  Wt[i] = (n < Nc && k < K) ? (_Float16)W[(size_t)k * Nc + n] : (_Float16)0.f;
}

// ------------------------------ WMMA GEMM ----------------------------------
// C[M, Cstride](f32) = A[M, Kp](f16) * Bt[Ncols, Kp](f16, col-major weights)
// One 16x16 tile per wave; 4 waves per block; grid = (ceil(M/64), Ncols/16).
__global__ __launch_bounds__(128)
void wmma_gemm_f16(const _Float16* __restrict__ A, const _Float16* __restrict__ Bt,
                   float* __restrict__ C, int M, int Kp, int Cstride) {
  const int lane  = threadIdx.x & 31;
  const int wave  = threadIdx.x >> 5;
  const int tileM = blockIdx.x * 4 + wave;
  const int row0  = tileM * 16;
  if (row0 >= M) return;                         // uniform per wave -> EXEC stays all-1
  const int ncol0 = blockIdx.y * 16;
  const int lrow  = lane & 15;
  const int lhalf = lane >> 4;

  const _Float16* __restrict__ Arow = A  + (size_t)(row0 + lrow) * Kp;
  const _Float16* __restrict__ Bcol = Bt + (size_t)(ncol0 + lrow) * Kp;

  v8f acc = {};
  for (int kb = 0; kb < Kp; kb += 32) {
    v16h a, b;
#pragma unroll
    for (int v = 0; v < 8; ++v) {
      // A 16x32 f16 layout: v0-3 hold K {2v,2v+1}+8*lhalf ; v4-7 hold K {16+2(v-4),..}+8*lhalf
      int ak = ((v < 4) ? (2 * v) : (16 + 2 * (v - 4))) + lhalf * 8;
      a[2 * v]     = Arow[kb + ak];
      a[2 * v + 1] = Arow[kb + ak + 1];
      // B 32x16 f16 layout: lane = column, same K mapping as A (v0-3: {2v,2v+1}+8*lhalf; v4-7: {16+2(v-4),..}+8*lhalf)
      int bk = ((v < 4) ? (2 * v) : (16 + 2 * (v - 4))) + lhalf * 8;
      b[2 * v]     = Bcol[kb + bk];
      b[2 * v + 1] = Bcol[kb + bk + 1];
    }
    acc = __builtin_amdgcn_wmma_f32_16x16x32_f16(false, a, false, b,
                                                 (short)0, acc, false, false);
  }
  // D layout: VGPR r -> row r + 8*lhalf, col = lane&15
  float* __restrict__ Crow = C + (size_t)(row0 + lhalf * 8) * Cstride + ncol0 + lrow;
#pragma unroll
  for (int r = 0; r < 8; ++r) Crow[(size_t)r * Cstride] = acc[r];
}

// --------------------------- attention -------------------------------------
__global__ void alpha_kernel(const float* __restrict__ h, int F, int H, int C,
                             const float* __restrict__ a_s, const float* __restrict__ a_d,
                             float* __restrict__ als, float* __restrict__ ald, int N) {
  int i = blockIdx.x * blockDim.x + threadIdx.x;
  if (i >= N * H) return;
  int n = i / H, hh = i % H;
  const float* hr = h + (size_t)n * F + hh * C;
  float s = 0.f, d = 0.f;
  for (int c = 0; c < C; ++c) { s += hr[c] * a_s[hh * C + c]; d += hr[c] * a_d[hh * C + c]; }
  als[i] = s; ald[i] = d;
}

__global__ void edge_max_kernel(const int* __restrict__ ei, int E, int ET, int H,
                                const float* __restrict__ als, const float* __restrict__ ald,
                                float* __restrict__ emax) {
  int i = blockIdx.x * blockDim.x + threadIdx.x;
  if (i >= ET * H) return;
  int e = i / H, hh = i % H, s, d;
  edge_endpoints(ei, E, e, s, d);
  float v = lrelu(als[s * H + hh] + ald[d * H + hh]);
  atomicMaxF(&emax[d * H + hh], v);
}

__global__ void edge_sum_kernel(const int* __restrict__ ei, int E, int ET, int H,
                                const float* __restrict__ als, const float* __restrict__ ald,
                                const float* __restrict__ emax, float* __restrict__ denom) {
  int i = blockIdx.x * blockDim.x + threadIdx.x;
  if (i >= ET * H) return;
  int e = i / H, hh = i % H, s, d;
  edge_endpoints(ei, E, e, s, d);
  float v = lrelu(als[s * H + hh] + ald[d * H + hh]);
  atomicAdd(&denom[d * H + hh], __expf(v - emax[d * H + hh]));
}

__global__ void edge_aggregate_kernel(const int* __restrict__ ei, int E, int ET,
                                      int H, int C, int F,
                                      const float* __restrict__ als, const float* __restrict__ ald,
                                      const float* __restrict__ emax, const float* __restrict__ denom,
                                      const float* __restrict__ h, float* __restrict__ agg) {
  int i = blockIdx.x * blockDim.x + threadIdx.x;
  if (i >= ET * H) return;
  int e = i / H, hh = i % H, s, d;
  edge_endpoints(ei, E, e, s, d);
  float v = lrelu(als[s * H + hh] + ald[d * H + hh]);
  float alpha = __expf(v - emax[d * H + hh]) / (denom[d * H + hh] + 1e-16f);
  const float* hs = h + (size_t)s * F + hh * C;
  float* ag = agg + (size_t)d * F + hh * C;
  for (int c = 0; c < C; ++c) atomicAdd(&ag[c], hs[c] * alpha);
}

// ----------------------------- BatchNorm -----------------------------------
__global__ __launch_bounds__(256)
void bn_stats_kernel(const float* __restrict__ h, int stride, int N,
                     float* __restrict__ sum, float* __restrict__ sumsq) {
  __shared__ float s1[256], s2[256];
  const int f = blockIdx.y;
  float p1 = 0.f, p2 = 0.f;
  for (int i = blockIdx.x * blockDim.x + threadIdx.x; i < N; i += gridDim.x * blockDim.x) {
    float v = h[(size_t)i * stride + f];
    p1 += v; p2 += v * v;
  }
  s1[threadIdx.x] = p1; s2[threadIdx.x] = p2;
  __syncthreads();
  for (int o = 128; o > 0; o >>= 1) {
    if ((int)threadIdx.x < o) { s1[threadIdx.x] += s1[threadIdx.x + o]; s2[threadIdx.x] += s2[threadIdx.x + o]; }
    __syncthreads();
  }
  if (threadIdx.x == 0) { atomicAdd(&sum[f], s1[0]); atomicAdd(&sumsq[f], s2[0]); }
}

__global__ void bn_apply_relu_kernel(float* __restrict__ h, int stride, int F, int N,
                                     const float* __restrict__ sum, const float* __restrict__ sumsq,
                                     const float* __restrict__ gamma, const float* __restrict__ beta) {
  int i = blockIdx.x * blockDim.x + threadIdx.x;
  if (i >= N * F) return;
  int n = i / F, f = i % F;
  float fn = (float)N;
  float m  = sum[f] / fn;
  float vv = sumsq[f] / fn - m * m;
  float x  = h[(size_t)n * stride + f];
  float y  = gamma[f] * (x - m) * rsqrtf(vv + EPS_BN) + beta[f];
  h[(size_t)n * stride + f] = fmaxf(y, 0.f);
}

// ------------------------- pool + classifier -------------------------------
__global__ void count_kernel(const int* __restrict__ batch, int N, float* __restrict__ counts) {
  int i = blockIdx.x * blockDim.x + threadIdx.x;
  if (i < N) atomicAdd(&counts[batch[i]], 1.0f);
}

__global__ void pool_kernel(const float* __restrict__ h, const int* __restrict__ batch,
                            int N, int F, float* __restrict__ pooled) {
  int i = blockIdx.x * blockDim.x + threadIdx.x;
  if (i >= N * F) return;
  int n = i / F, f = i % F;
  atomicAdd(&pooled[(size_t)batch[n] * F + f], h[i]);
}

__global__ void classifier_kernel(const float* __restrict__ pooled, const float* __restrict__ counts,
                                  const float* __restrict__ Wo, const float* __restrict__ bo,
                                  float* __restrict__ out, int G, int NC, int F) {
  int i = blockIdx.x * blockDim.x + threadIdx.x;
  if (i >= G * NC) return;
  int g = i / NC, c = i % NC;
  float inv = 1.0f / fmaxf(counts[g], 1.0f);
  float s = bo[c];
  for (int k = 0; k < F; ++k) s += pooled[(size_t)g * F + k] * inv * Wo[k * NC + c];
  out[i] = s;
}

// ---------------------------------------------------------------------------
static inline int cdiv(int a, int b) { return (a + b - 1) / b; }

extern "C" void kernel_launch(void* const* d_in, const int* in_sizes, int n_in,
                              void* d_out, int out_size, void* d_ws, size_t ws_size,
                              hipStream_t stream) {
  const int N = 100000, E = 1000000, ET = E + N, G = 512, NC = 10;
  const int TB = 256;
  const int tilesM = N / 16;                 // 6250, exact
  const int gx = cdiv(tilesM, 4);            // waves-per-block = 4

  // inputs (setup_inputs order)
  const float* x   = (const float*)d_in[0];
  const int*   ei  = (const int*)d_in[1];
  const int*   bat = (const int*)d_in[2];
  const float* W1  = (const float*)d_in[3];
  const float* as1 = (const float*)d_in[4];
  const float* ad1 = (const float*)d_in[5];
  const float* b1  = (const float*)d_in[6];
  const float* g1  = (const float*)d_in[7];
  const float* be1 = (const float*)d_in[8];
  const float* Wl1 = (const float*)d_in[9];
  const float* gl1 = (const float*)d_in[11];
  const float* bel1= (const float*)d_in[12];
  const float* W2  = (const float*)d_in[13];
  const float* as2 = (const float*)d_in[14];
  const float* ad2 = (const float*)d_in[15];
  const float* b2  = (const float*)d_in[16];
  const float* g2  = (const float*)d_in[17];
  const float* be2 = (const float*)d_in[18];
  const float* Wl2 = (const float*)d_in[19];
  const float* gl2 = (const float*)d_in[21];
  const float* bel2= (const float*)d_in[22];
  const float* W3  = (const float*)d_in[23];
  const float* as3 = (const float*)d_in[24];
  const float* ad3 = (const float*)d_in[25];
  const float* b3  = (const float*)d_in[26];
  const float* g3  = (const float*)d_in[27];
  const float* be3 = (const float*)d_in[28];
  const float* Wo  = (const float*)d_in[29];
  const float* bo  = (const float*)d_in[30];
  // note: bl1/bl2 (d_in[10]/d_in[20]) are immediately followed by BN -> exact cancel.
  float* out = (float*)d_out;

  // workspace carve-out
  size_t off = 0;
  auto carve = [&](size_t bytes) -> void* {
    off = (off + 255) & ~(size_t)255;
    void* p = (char*)d_ws + off;
    off += bytes;
    return p;
  };
  _Float16* Ah   = (_Float16*)carve((size_t)N * 128 * sizeof(_Float16)); // staged A
  _Float16* Wt   = (_Float16*)carve((size_t)128 * 128 * sizeof(_Float16));
  float*    Hg   = (float*)carve((size_t)N * 128 * sizeof(float));       // GEMM out / linear act
  float*    AGG  = (float*)carve((size_t)N * 128 * sizeof(float));       // aggregation / conv act
  float*    als  = (float*)carve((size_t)N * 8 * sizeof(float));
  float*    ald  = (float*)carve((size_t)N * 8 * sizeof(float));
  float*    emax = (float*)carve((size_t)N * 8 * sizeof(float));
  float*    den  = (float*)carve((size_t)N * 8 * sizeof(float));
  float*    stats= (float*)carve(256 * sizeof(float));                   // sum | sumsq
  float*    pool = (float*)carve((size_t)G * 64 * sizeof(float));
  float*    cnts = (float*)carve((size_t)G * sizeof(float));
  (void)in_sizes; (void)n_in; (void)out_size; (void)ws_size;

  auto run_bn = [&](float* h, int stride, int F, const float* gm, const float* bt) {
    fill_kernel<<<1, TB, 0, stream>>>(stats, 0.f, 2 * F);
    bn_stats_kernel<<<dim3(32, F), TB, 0, stream>>>(h, stride, N, stats, stats + F);
    bn_apply_relu_kernel<<<cdiv(N * F, TB), TB, 0, stream>>>(h, stride, F, N,
                                                             stats, stats + F, gm, bt);
  };

  auto run_attention = [&](const float* h, int F, int H, int C,
                           const float* a_s, const float* a_d, const float* bias) {
    alpha_kernel<<<cdiv(N * H, TB), TB, 0, stream>>>(h, F, H, C, a_s, a_d, als, ald, N);
    fill_kernel<<<cdiv(N * H, TB), TB, 0, stream>>>(emax, -3.0e38f, N * H);
    fill_kernel<<<cdiv(N * H, TB), TB, 0, stream>>>(den, 0.f, N * H);
    edge_max_kernel<<<cdiv(ET * H, TB), TB, 0, stream>>>(ei, E, ET, H, als, ald, emax);
    edge_sum_kernel<<<cdiv(ET * H, TB), TB, 0, stream>>>(ei, E, ET, H, als, ald, emax, den);
    fill_bias_kernel<<<cdiv(N * F, TB), TB, 0, stream>>>(AGG, bias, F, N * F);
    edge_aggregate_kernel<<<cdiv(ET * H, TB), TB, 0, stream>>>(ei, E, ET, H, C, F,
                                                               als, ald, emax, den, h, AGG);
  };

  // ---------------- conv1: x[ N,128 ] @ W1[128,128], H=8, C=16 -------------
  conv_a_f16<<<cdiv(N * 128, TB), TB, 0, stream>>>(x, 128, 128, Ah, 128, N);
  conv_w_f16<<<cdiv(128 * 128, TB), TB, 0, stream>>>(W1, 128, 128, Wt, 128, 128);
  wmma_gemm_f16<<<dim3(gx, 8), 128, 0, stream>>>(Ah, Wt, Hg, N, 128, 128);
  run_attention(Hg, 128, 8, 16, as1, ad1, b1);
  run_bn(AGG, 128, 128, g1, be1);

  // ---------------- lin1: act[N,128] @ Wl1[128,16] -------------------------
  conv_a_f16<<<cdiv(N * 128, TB), TB, 0, stream>>>(AGG, 128, 128, Ah, 128, N);
  conv_w_f16<<<cdiv(16 * 128, TB), TB, 0, stream>>>(Wl1, 128, 16, Wt, 128, 16);
  wmma_gemm_f16<<<dim3(gx, 1), 128, 0, stream>>>(Ah, Wt, Hg, N, 128, 16);
  run_bn(Hg, 16, 16, gl1, bel1);

  // ---------------- conv2: act[N,16] @ W2[16,96], H=4, C=24 (K pad 32) -----
  conv_a_f16<<<cdiv(N * 32, TB), TB, 0, stream>>>(Hg, 16, 16, Ah, 32, N);
  conv_w_f16<<<cdiv(96 * 32, TB), TB, 0, stream>>>(W2, 16, 96, Wt, 32, 96);
  wmma_gemm_f16<<<dim3(gx, 6), 128, 0, stream>>>(Ah, Wt, Hg, N, 32, 96);
  run_attention(Hg, 96, 4, 24, as2, ad2, b2);
  run_bn(AGG, 96, 96, g2, be2);

  // ---------------- lin2: act[N,96] @ Wl2[96,24] (N pad 32) ----------------
  conv_a_f16<<<cdiv(N * 96, TB), TB, 0, stream>>>(AGG, 96, 96, Ah, 96, N);
  conv_w_f16<<<cdiv(32 * 96, TB), TB, 0, stream>>>(Wl2, 96, 24, Wt, 96, 32);
  wmma_gemm_f16<<<dim3(gx, 2), 128, 0, stream>>>(Ah, Wt, Hg, N, 96, 32);  // Cstride=32
  run_bn(Hg, 32, 24, gl2, bel2);

  // ---------------- conv3: act[N,24] @ W3[24,64], H=2, C=32 (K pad 32) -----
  conv_a_f16<<<cdiv(N * 32, TB), TB, 0, stream>>>(Hg, 32, 24, Ah, 32, N);
  conv_w_f16<<<cdiv(64 * 32, TB), TB, 0, stream>>>(W3, 24, 64, Wt, 32, 64);
  wmma_gemm_f16<<<dim3(gx, 4), 128, 0, stream>>>(Ah, Wt, Hg, N, 32, 64);
  run_attention(Hg, 64, 2, 32, as3, ad3, b3);
  run_bn(AGG, 64, 64, g3, be3);

  // ---------------- mean pool + classifier ---------------------------------
  fill_kernel<<<cdiv(G * 64, TB), TB, 0, stream>>>(pool, 0.f, G * 64);
  fill_kernel<<<cdiv(G, TB), TB, 0, stream>>>(cnts, 0.f, G);
  count_kernel<<<cdiv(N, TB), TB, 0, stream>>>(bat, N, cnts);
  pool_kernel<<<cdiv(N * 64, TB), TB, 0, stream>>>(AGG, bat, N, 64, pool);
  classifier_kernel<<<cdiv(G * NC, TB), TB, 0, stream>>>(pool, cnts, Wo, bo, out, G, NC, 64);
}
